// GRU_56607668961499
// MI455X (gfx1250) — matrix operand order repaired and checked by reading, not compiled, not run on hardware
//
#include <hip/hip_runtime.h>
#include <stdint.h>
#include <stddef.h>

typedef __attribute__((ext_vector_type(16))) __bf16    v16bf;
typedef __attribute__((ext_vector_type(8)))  float     v8f;
typedef __attribute__((ext_vector_type(4)))  uint32_t  u32x4;

#define B_    128
#define T_    512
#define PRED_ 24
#define F_    16
#define L_    8
#define H_    1024
#define G3H   (3 * H_)
#define FTOT  (T_ + PRED_ - 1)   // 535 feature timesteps
#define BPITCH 40                // LDS row pitch in ushorts (32 data + 8 pad)

// ---------- scalar helpers ----------
__device__ __forceinline__ unsigned short f2bf(float f) {
  uint32_t u = __float_as_uint(f);
  u += 0x7fffu + ((u >> 16) & 1u);   // round-to-nearest-even
  return (unsigned short)(u >> 16);
}
__device__ __forceinline__ float bf2f(unsigned short s) {
  return __uint_as_float(((uint32_t)s) << 16);
}
__device__ __forceinline__ float sigmoidf_(float x) {
  return 1.0f / (1.0f + __expf(-x));
}

union FragAB { u32x4 q[2]; v16bf v; };

// ---------- small prep kernels ----------
__global__ void conv_pad_bf16(const float* __restrict__ src, unsigned short* __restrict__ dst,
                              int rows, int scols, int dcols) {
  int i = blockIdx.x * blockDim.x + threadIdx.x;
  if (i >= rows * dcols) return;
  int r = i / dcols, c = i - r * dcols;
  dst[i] = (c < scols) ? f2bf(src[(size_t)r * scols + c]) : (unsigned short)0;
}

__global__ void zero_bf16(unsigned short* __restrict__ p, int n) {
  int i = blockIdx.x * blockDim.x + threadIdx.x;
  if (i < n) p[i] = 0;
}

// xcat[b,t,0:16]=feats, [16:24]=labels, [24:32]=0   (K padded 24 -> 32)
__global__ void build_xcat(const float* __restrict__ feats, const float* __restrict__ labels,
                           unsigned short* __restrict__ xcat) {
  int i = blockIdx.x * blockDim.x + threadIdx.x;
  if (i >= B_ * T_ * 32) return;
  int c = i & 31, bt = i >> 5, t = bt % T_, b = bt / T_;
  float v = 0.0f;
  if (c < F_)            v = feats[((size_t)b * FTOT + t) * F_ + c];
  else if (c < F_ + L_)  v = labels[((size_t)b * T_ + t) * L_ + (c - F_)];
  xcat[i] = f2bf(v);
}

// decoder input: xin[b,0:16]=feats[b, T+step, :], [16:24]=p_prev(bf16), [24:32]=0
__global__ void build_xin(const float* __restrict__ feats, const unsigned short* __restrict__ p,
                          unsigned short* __restrict__ xin, int stepi) {
  int i = blockIdx.x * blockDim.x + threadIdx.x;
  if (i >= B_ * 32) return;
  int c = i & 31, b = i >> 5;
  unsigned short v = 0;
  if (c < F_)            v = f2bf(feats[((size_t)b * FTOT + T_ + stepi) * F_ + c]);
  else if (c < F_ + L_)  v = p[b * L_ + (c - F_)];
  xin[i] = v;
}

// out[b,j] = h(bf16) . fcW[j,:] + fcb[j]   (also emit bf16 copy for next decoder input)
__global__ void fc_head(const unsigned short* __restrict__ h, const float* __restrict__ W,
                        const float* __restrict__ bias, float* __restrict__ out,
                        unsigned short* __restrict__ p) {
  int i = blockIdx.x * blockDim.x + threadIdx.x;
  if (i >= B_ * L_) return;
  int b = i >> 3, j = i & 7;
  float acc = bias[j];
  const unsigned short* hr = h + (size_t)b * H_;
  const float* wr = W + (size_t)j * H_;
  for (int k = 0; k < H_; ++k) acc += bf2f(hr[k]) * wr[k];
  out[i] = acc;
  p[i] = f2bf(acc);
}

// ---------- pipelined GEMM phase (one K region, fixed accumulators) ----------
// Computes, for NITER k-tiles of 32:
//   acc0 += A x W[rows g=0]   (r gate)
//   acc1 += A x W[rows g=1]   (z gate)
//   acc2 += A x W[rows g=2]   (n gate: i_n or h_n depending on caller)
// B tiles staged through double-buffered LDS; one barrier per iteration.
template<int NITER>
__device__ __forceinline__ void gemm_phase(
    const unsigned short* __restrict__ W, int kst,            // (3H, kst) bf16 row-major
    const unsigned short* __restrict__ a_base, int astride,   // A rows (128, >=NITER*32)
    unsigned short (&lds)[2][3][32][BPITCH],
    int tid, int lane, int wave, int nbase,
    v8f (&acc0)[2], v8f (&acc1)[2], v8f (&acc2)[2]) {

  const int mrow = wave * 16 + (lane & 15);   // batch row for A fragment
  const int akb  = (lane >> 4) << 4;          // A: K offset 0/16 (16 contiguous)
  const int bks  = (lane >> 4) << 4;          // B: K offset 0/16 (16 contiguous)

  // static chunk split: 384 16-byte chunks, thread does {tid} and maybe {tid+256}
  const int  g0 = tid >> 7, r0 = (tid >> 2) & 31, c0 = (tid & 3) << 3;
  const bool has1 = (tid < 128);
  const int  t1 = tid + 256;
  const int  g1 = t1 >> 7, r1 = (t1 >> 2) & 31, c1 = (t1 & 3) << 3;

  const unsigned short* src0 = W + (size_t)(g0 * H_ + nbase + r0) * kst + c0;
  const unsigned short* src1 = W + (size_t)(g1 * H_ + nbase + r1) * kst + c1;

  __syncthreads();   // protect LDS buffer 0 against readers from a previous phase

  // prologue: stage tile 0 into buffer 0
  u32x4 p0 = *(const u32x4*)(src0);
  u32x4 p1;
  if (has1) p1 = *(const u32x4*)(src1);
  *(u32x4*)(&lds[0][g0][r0][c0]) = p0;
  if (has1) *(u32x4*)(&lds[0][g1][r1][c1]) = p1;
  __syncthreads();

  for (int it = 0; it < NITER; ++it) {
    const int cur = it & 1;
    const int nxt = cur ^ 1;

    // issue global loads for next tile (overlaps with WMMA below)
    if (it + 1 < NITER) {
      p0 = *(const u32x4*)(src0 + (it + 1) * 32);
      if (has1) p1 = *(const u32x4*)(src1 + (it + 1) * 32);
    }

    // A fragment (16x32, contiguous k: lane half h holds k 16h..16h+15)
    FragAB af;
    const unsigned short* arow = a_base + (size_t)mrow * astride + it * 32;
    af.q[0] = *(const u32x4*)(arow + akb);
    af.q[1] = *(const u32x4*)(arow + akb + 8);

#pragma unroll
    for (int j = 0; j < 2; ++j) {
      const int nl = j * 16 + (lane & 15);
      FragAB br, bz, bn;
      br.q[0] = *(const u32x4*)(&lds[cur][0][nl][bks]);
      br.q[1] = *(const u32x4*)(&lds[cur][0][nl][bks + 8]);
      bz.q[0] = *(const u32x4*)(&lds[cur][1][nl][bks]);
      bz.q[1] = *(const u32x4*)(&lds[cur][1][nl][bks + 8]);
      bn.q[0] = *(const u32x4*)(&lds[cur][2][nl][bks]);
      bn.q[1] = *(const u32x4*)(&lds[cur][2][nl][bks + 8]);

      acc0[j] = __builtin_amdgcn_wmma_f32_16x16x32_bf16(
          false, af.v, false, br.v, (short)0, acc0[j], false, false);
      acc1[j] = __builtin_amdgcn_wmma_f32_16x16x32_bf16(
          false, af.v, false, bz.v, (short)0, acc1[j], false, false);
      acc2[j] = __builtin_amdgcn_wmma_f32_16x16x32_bf16(
          false, af.v, false, bn.v, (short)0, acc2[j], false, false);
    }

    // stage next tile, single barrier per iteration
    if (it + 1 < NITER) {
      *(u32x4*)(&lds[nxt][g0][r0][c0]) = p0;
      if (has1) *(u32x4*)(&lds[nxt][g1][r1][c1]) = p1;
      __syncthreads();
    }
  }
}

// ---------- fused GRU step: gates GEMM (WMMA bf16) + GRU nonlinearity ----------
// grid = 32 blocks (each owns a 32-wide hidden slice, all 3 gate groups)
// block = 256 threads = 8 waves; wave w computes batch rows 16w..16w+15
template<int KX>
__global__ __launch_bounds__(256)
void gru_step(const unsigned short* __restrict__ x, int xstride,
              const unsigned short* __restrict__ hin, unsigned short* __restrict__ hout,
              const unsigned short* __restrict__ Wih,  // (3H, KX) bf16 row-major
              const unsigned short* __restrict__ Whh,  // (3H, H)  bf16 row-major
              const float* __restrict__ bih, const float* __restrict__ bhh) {
  __shared__ __align__(16) unsigned short ldsB[2][3][32][BPITCH];

  const int tid   = threadIdx.x;
  const int lane  = tid & 31;
  const int wave  = tid >> 5;
  const int nbase = blockIdx.x * 32;

  v8f acc_r[2], acc_z[2], acc_in[2], acc_hn[2];
#pragma unroll
  for (int j = 0; j < 2; ++j) {
    acc_r[j] = (v8f)0.0f; acc_z[j] = (v8f)0.0f;
    acc_in[j] = (v8f)0.0f; acc_hn[j] = (v8f)0.0f;
  }

  // Phase 1: input region (gi contributions): r, z, i_n
  gemm_phase<KX / 32>(Wih, KX, x, xstride, ldsB, tid, lane, wave, nbase,
                      acc_r, acc_z, acc_in);
  // Phase 2: hidden region (gh contributions): r, z, h_n
  gemm_phase<H_ / 32>(Whh, H_, hin, H_, ldsB, tid, lane, wave, nbase,
                      acc_r, acc_z, acc_hn);

  // GRU epilogue — C layout: row = v + 8*(lane>=16), col = lane&15
#pragma unroll
  for (int j = 0; j < 2; ++j) {
    const int n_g = nbase + j * 16 + (lane & 15);
    const float brz = bih[n_g] + bhh[n_g];
    const float bzz = bih[H_ + n_g] + bhh[H_ + n_g];
    const float bin = bih[2 * H_ + n_g];
    const float bhn = bhh[2 * H_ + n_g];
#pragma unroll
    for (int v = 0; v < 8; ++v) {
      const int m = wave * 16 + v + ((lane >> 4) << 3);
      const float r = sigmoidf_(acc_r[j][v] + brz);
      const float z = sigmoidf_(acc_z[j][v] + bzz);
      const float n = tanhf(acc_in[j][v] + bin + r * (acc_hn[j][v] + bhn));
      const float hold = bf2f(hin[(size_t)m * H_ + n_g]);
      hout[(size_t)m * H_ + n_g] = f2bf((1.0f - z) * n + z * hold);
    }
  }
}

// ---------- host ----------
extern "C" void kernel_launch(void* const* d_in, const int* in_sizes, int n_in,
                              void* d_out, int out_size, void* d_ws, size_t ws_size,
                              hipStream_t stream) {
  (void)in_sizes; (void)n_in; (void)out_size; (void)ws_size;
  const float* feats  = (const float*)d_in[0];
  const float* labels = (const float*)d_in[1];
  // d_in[2]=y, d_in[3]=teacher : teacher==0 path (autoregressive) implemented
  const float* Wih0 = (const float*)d_in[4];
  const float* Whh0 = (const float*)d_in[5];
  const float* bih0 = (const float*)d_in[6];
  const float* bhh0 = (const float*)d_in[7];
  const float* Wih1 = (const float*)d_in[8];
  const float* Whh1 = (const float*)d_in[9];
  const float* bih1 = (const float*)d_in[10];
  const float* bhh1 = (const float*)d_in[11];
  const float* dWih = (const float*)d_in[12];
  const float* dWhh = (const float*)d_in[13];
  const float* dbih = (const float*)d_in[14];
  const float* dbhh = (const float*)d_in[15];
  const float* fcW  = (const float*)d_in[16];
  const float* fcb  = (const float*)d_in[17];
  float* out = (float*)d_out;

  char* ws = (char*)d_ws;
  size_t off = 0;
  auto alloc = [&](size_t bytes) -> void* {
    off = (off + 255) & ~(size_t)255;
    void* p = ws + off;
    off += bytes;
    return p;
  };

  unsigned short* wih0p = (unsigned short*)alloc((size_t)G3H * 32 * 2);
  unsigned short* whh0b = (unsigned short*)alloc((size_t)G3H * H_ * 2);
  unsigned short* wih1b = (unsigned short*)alloc((size_t)G3H * H_ * 2);
  unsigned short* whh1b = (unsigned short*)alloc((size_t)G3H * H_ * 2);
  unsigned short* dwihp = (unsigned short*)alloc((size_t)G3H * 32 * 2);
  unsigned short* dwhhb = (unsigned short*)alloc((size_t)G3H * H_ * 2);
  unsigned short* xcat  = (unsigned short*)alloc((size_t)B_ * T_ * 32 * 2);
  unsigned short* xin   = (unsigned short*)alloc((size_t)B_ * 32 * 2);
  unsigned short* h0[2] = {(unsigned short*)alloc((size_t)B_ * H_ * 2),
                           (unsigned short*)alloc((size_t)B_ * H_ * 2)};
  unsigned short* h1[2] = {(unsigned short*)alloc((size_t)B_ * H_ * 2),
                           (unsigned short*)alloc((size_t)B_ * H_ * 2)};
  unsigned short* pbuf  = (unsigned short*)alloc((size_t)B_ * L_ * 2);

  auto blocks = [](int n) { return (n + 255) / 256; };

  // one-time (per call) conversions
  conv_pad_bf16<<<blocks(G3H * 32), 256, 0, stream>>>(Wih0, wih0p, G3H, F_ + L_, 32);
  conv_pad_bf16<<<blocks(G3H * H_), 256, 0, stream>>>(Whh0, whh0b, G3H, H_, H_);
  conv_pad_bf16<<<blocks(G3H * H_), 256, 0, stream>>>(Wih1, wih1b, G3H, H_, H_);
  conv_pad_bf16<<<blocks(G3H * H_), 256, 0, stream>>>(Whh1, whh1b, G3H, H_, H_);
  conv_pad_bf16<<<blocks(G3H * 32), 256, 0, stream>>>(dWih, dwihp, G3H, F_ + L_, 32);
  conv_pad_bf16<<<blocks(G3H * H_), 256, 0, stream>>>(dWhh, dwhhb, G3H, H_, H_);
  build_xcat<<<blocks(B_ * T_ * 32), 256, 0, stream>>>(feats, labels, xcat);
  zero_bf16<<<blocks(B_ * H_), 256, 0, stream>>>(h0[0], B_ * H_);
  zero_bf16<<<blocks(B_ * H_), 256, 0, stream>>>(h1[0], B_ * H_);

  // encoder: 512 sequential steps, two fused layer-step kernels each
  for (int t = 0; t < T_; ++t) {
    gru_step<32><<<32, 256, 0, stream>>>(xcat + (size_t)t * 32, T_ * 32,
                                         h0[t & 1], h0[(t + 1) & 1],
                                         wih0p, whh0b, bih0, bhh0);
    gru_step<H_><<<32, 256, 0, stream>>>(h0[(t + 1) & 1], H_,
                                         h1[t & 1], h1[(t + 1) & 1],
                                         wih1b, whh1b, bih1, bhh1);
  }

  // ps = head(h_enc)  (encoder final state is in h1[0] since T is even)
  fc_head<<<4, 256, 0, stream>>>(h1[0], fcW, fcb, out, pbuf);

  // autoregressive decoder: 23 steps
  for (int i = 0; i < PRED_ - 1; ++i) {
    build_xin<<<16, 256, 0, stream>>>(feats, pbuf, xin, i);
    gru_step<32><<<32, 256, 0, stream>>>(xin, 32,
                                         h1[i & 1], h1[(i + 1) & 1],
                                         dwihp, dwhhb, dbih, dbhh);
    fc_head<<<4, 256, 0, stream>>>(h1[(i + 1) & 1], fcW, fcb,
                                   out + (size_t)(i + 1) * B_ * L_, pbuf);
  }
}
